// GCRN_35012573397316
// MI455X (gfx1250) — matrix operand order repaired and checked by reading, not compiled, not run on hardware
//
#include <hip/hip_runtime.h>
#include <hip/hip_bf16.h>
#include <math.h>

// ---------------------------------------------------------------------------
// GCRN (adaptive-graph GRU) for MI455X / gfx1250.
// Dominant cost: dense graph-diffusion GEMMs (S[1024x1024] @ act[1024 x B*C]),
// ~1 TFLOP/call, working set < L2 (192MB)  => compute-bound => bf16 WMMA
// (v_wmma_f32_16x16x32_bf16, fp32 accumulate).
// 64x64 register-blocked tiles (4x4 fragments / wave, 32 FLOP/B of L2
// traffic); __launch_bounds__(256,2) lifts the VGPR cap so the 16 accumulator
// fragments live entirely in registers (no scratch spills).
// ---------------------------------------------------------------------------

typedef __bf16 bf16_t;
typedef __attribute__((ext_vector_type(16))) __bf16        v16bf;
typedef __attribute__((ext_vector_type(8)))  float         v8f;
typedef __attribute__((ext_vector_type(8)))  unsigned int  v8u;
typedef __attribute__((ext_vector_type(4)))  unsigned int  v4u;

#define NN   1024           // nodes
#define BB   64             // batch
#define TT   12             // encoder steps
#define HZ   12             // horizon
#define HH   64             // hidden
#define CENC 65             // DIN + H
#define CDEC 66             // DOUT + YCOV + H
#define JENC (BB*CENC)      // 4160
#define JDEC (BB*CDEC)      // 4224
#define KP   224            // padded reduction dim for dense GEMMs (7*32)
#define RTOT (NN*BB)        // 65536 rows (n-major, b-minor)

union Frag { v8u u; v4u q[2]; v16bf b; };

__device__ __forceinline__ bf16_t tobf(float f) {
  unsigned u = __float_as_uint(f);
  u += 0x7FFFu + ((u >> 16) & 1u);              // round-to-nearest-even
  unsigned short s = (unsigned short)(u >> 16);
  return __builtin_bit_cast(bf16_t, s);
}

// A fragment, 16x32 bf16, row-major source.
// lane&15 = row; lane>>4 selects K half {0..15} vs {16..31}.
__device__ __forceinline__ v16bf load_afrag(const bf16_t* rowp, int hi) {
  const v4u* p = (const v4u*)rowp;      // rowp = &A[row][k0], 16B-aligned
  Frag f;
  f.q[0] = p[2 * hi];                    // k = 16*hi .. 16*hi+7
  f.q[1] = p[2 * hi + 1];                // k = 16*hi+8 .. 16*hi+15
  return f.b;
}

// B fragment, 32x16 bf16, from a K-contiguous (transposed) source.
// lane&15 = col; lane>>4 selects K half {0..15} vs {16..31}.
// colp = &BT[col][k0 + 16*hi]  (contiguous 32 bytes)
__device__ __forceinline__ v16bf load_bfrag(const bf16_t* colp) {
  const v4u* p = (const v4u*)colp;
  Frag f;
  f.q[0] = p[0];
  f.q[1] = p[1];
  return f.b;
}

__device__ __forceinline__ float sigmoidf_(float v) {
  return 1.0f / (1.0f + __expf(-v));
}

// ---------------------------------------------------------------------------
// Register-blocked WMMA core: one wave computes a 64x64 tile as 4x4 fragments.
// A: row-major [.,lda] (reduction contiguous); BT: col-major-as-rows [.,ldb]
// (reduction contiguous). acc must be zero-initialized by caller.
// Loop order keeps peak liveness ~= acc(128) + bfr(32) + af(8) VGPRs.
// ---------------------------------------------------------------------------
__device__ __forceinline__ void mma_tile64(const bf16_t* __restrict__ A,
                                           size_t lda, int n0,
                                           const bf16_t* __restrict__ BT,
                                           size_t ldb, int j0,
                                           int klen, int lo, int hi,
                                           v8f acc[4][4]) {
  const bf16_t* ap[4];
  const bf16_t* bp[4];
#pragma unroll
  for (int r = 0; r < 4; ++r) ap[r] = A + (size_t)(n0 + 16 * r + lo) * lda;
#pragma unroll
  for (int s = 0; s < 4; ++s) bp[s] = BT + (size_t)(j0 + 16 * s + lo) * ldb + hi * 16;

  for (int k0 = 0; k0 < klen; k0 += 32) {
    v16bf bfr[4];
#pragma unroll
    for (int s = 0; s < 4; ++s) bfr[s] = load_bfrag(bp[s] + k0);
#pragma unroll
    for (int r = 0; r < 4; ++r) {
      v16bf af = load_afrag(ap[r] + k0, hi);
#pragma unroll
      for (int s = 0; s < 4; ++s)
        acc[r][s] = __builtin_amdgcn_wmma_f32_16x16x32_bf16(
            false, af, false, bfr[s], (short)0, acc[r][s], false, false);
    }
  }
}

// ---------------------------------------------------------------------------
// 1) Adjacency: S = softmax(relu(E E^T), axis=1); writes bf16 S and S^T.
// ---------------------------------------------------------------------------
__global__ void k_adj(const float* __restrict__ E,
                      bf16_t* __restrict__ Sb, bf16_t* __restrict__ STb) {
  __shared__ float row[NN];
  __shared__ float red[256];
  const int i = blockIdx.x, tid = threadIdx.x;
  float ei[8];
#pragma unroll
  for (int k = 0; k < 8; ++k) ei[k] = E[i * 8 + k];

  float mx = -1e30f;
  for (int j = tid; j < NN; j += 256) {
    float d = 0.f;
#pragma unroll
    for (int k = 0; k < 8; ++k) d += ei[k] * E[j * 8 + k];
    d = fmaxf(d, 0.f);                   // relu
    row[j] = d;
    mx = fmaxf(mx, d);
  }
  red[tid] = mx; __syncthreads();
  for (int s = 128; s > 0; s >>= 1) {
    if (tid < s) red[tid] = fmaxf(red[tid], red[tid + s]);
    __syncthreads();
  }
  const float m = red[0]; __syncthreads();

  float sum = 0.f;
  for (int j = tid; j < NN; j += 256) {
    float e = __expf(row[j] - m);
    row[j] = e; sum += e;
  }
  red[tid] = sum; __syncthreads();
  for (int s = 128; s > 0; s >>= 1) {
    if (tid < s) red[tid] += red[tid + s];
    __syncthreads();
  }
  const float inv = 1.0f / red[0];
  for (int j = tid; j < NN; j += 256) {
    bf16_t v = tobf(row[j] * inv);
    Sb [(size_t)i * NN + j] = v;
    STb[(size_t)j * NN + i] = v;         // transposed copy for WMMA B operand
  }
}

// ---------------------------------------------------------------------------
// 2) Chebyshev: S2 = 2*S@S - I  (bf16 WMMA, fp32 accumulate). 64x64 tiles.
// ---------------------------------------------------------------------------
__global__ void __launch_bounds__(256, 2)
k_cheb(const bf16_t* __restrict__ S, const bf16_t* __restrict__ ST,
       bf16_t* __restrict__ S2) {
  const int gw   = blockIdx.x * 8 + (threadIdx.x >> 5);
  const int lane = threadIdx.x & 31;
  const int lo = lane & 15, hi = lane >> 4;
  const int tn = gw >> 4, tj = gw & 15;
  const int n0 = tn * 64, j0 = tj * 64;
  v8f acc[4][4] = {};
  mma_tile64(S, NN, n0, ST, NN, j0, NN, lo, hi, acc);
#pragma unroll
  for (int r = 0; r < 4; ++r)
#pragma unroll
    for (int s = 0; s < 4; ++s) {
      const int col = j0 + 16 * s + lo;
#pragma unroll
      for (int g = 0; g < 8; ++g) {
        int n = n0 + 16 * r + g + 8 * hi;
        float v = 2.0f * acc[r][s][g] - (n == col ? 1.0f : 0.0f);
        S2[(size_t)n * NN + col] = tobf(v);
      }
    }
}

// ---------------------------------------------------------------------------
// 3) Weight pack: WT[o][q] = bf16(W[q][o]), q padded with zeros to KP=224.
// ---------------------------------------------------------------------------
__global__ void k_wpack(const float* __restrict__ W, bf16_t* __restrict__ WT,
                        int Krows, int O) {
  int idx = blockIdx.x * 256 + threadIdx.x;
  if (idx >= O * KP) return;
  int o = idx / KP, q = idx - o * KP;
  float v = (q < Krows) ? W[(size_t)q * O + o] : 0.f;
  WT[(size_t)o * KP + q] = tobf(v);
}

__global__ void k_zero(float* p, int nel) {
  int i = blockIdx.x * 256 + threadIdx.x;
  if (i < nel) p[i] = 0.f;
}

// ---------------------------------------------------------------------------
// 4) Build concat (encoder): c = [x_t, (z*)h]  ->  CT[(b*65+c)][n] (bf16, for
//    diffusion B operand) and XGCAT[(n*64+b)][0..64] (k=0 block) + zero pad.
// ---------------------------------------------------------------------------
__global__ void k_build_enc(const float* __restrict__ x, int t,
                            const float* __restrict__ h,
                            const float* __restrict__ zr, int usez,
                            bf16_t* __restrict__ CT, bf16_t* __restrict__ XG) {
  int r = blockIdx.x * 256 + threadIdx.x;
  if (r >= RTOT) return;
  int n = r >> 6, b = r & 63;
  float xv = x[((size_t)(b * TT + t)) * NN + n];
  bf16_t xb = tobf(xv);
  CT[((size_t)(b * CENC)) * NN + n] = xb;
  bf16_t* xrow = XG + (size_t)r * KP;
  xrow[0] = xb;
  for (int c = 0; c < HH; ++c) {
    float hv = h[(size_t)r * HH + c];
    if (usez) hv *= zr[(size_t)r * 128 + c];
    bf16_t hb = tobf(hv);
    CT[((size_t)(b * CENC + 1 + c)) * NN + n] = hb;
    xrow[1 + c] = hb;
  }
  for (int c = 3 * CENC; c < KP; ++c) xrow[c] = tobf(0.f);
}

// Decoder variant: c = [go, ycov_t, (z*)h], Cc = 66.
__global__ void k_build_dec(const float* __restrict__ go,
                            const float* __restrict__ ycov, int t,
                            const float* __restrict__ h,
                            const float* __restrict__ zr, int usez,
                            bf16_t* __restrict__ CT, bf16_t* __restrict__ XG) {
  int r = blockIdx.x * 256 + threadIdx.x;
  if (r >= RTOT) return;
  int n = r >> 6, b = r & 63;
  float gv = go[r];
  float yv = ycov[((size_t)(b * HZ + t)) * NN + n];
  bf16_t gb = tobf(gv), yb = tobf(yv);
  CT[((size_t)(b * CDEC + 0)) * NN + n] = gb;
  CT[((size_t)(b * CDEC + 1)) * NN + n] = yb;
  bf16_t* xrow = XG + (size_t)r * KP;
  xrow[0] = gb; xrow[1] = yb;
  for (int c = 0; c < HH; ++c) {
    float hv = h[(size_t)r * HH + c];
    if (usez) hv *= zr[(size_t)r * 128 + c];
    bf16_t hb = tobf(hv);
    CT[((size_t)(b * CDEC + 2 + c)) * NN + n] = hb;
    xrow[2 + c] = hb;
  }
  for (int c = 3 * CDEC; c < KP; ++c) xrow[c] = tobf(0.f);
}

// ---------------------------------------------------------------------------
// 5) Diffusion GEMM:  Y[n, j] = sum_m S[n,m] * CT[j][m]   (J = B*Cc)
//    64x64 tile / wave, K=1024. Result bf16 into XGCAT col block k*Cc.
//    Exact grids => EXEC all-ones at every WMMA.
// ---------------------------------------------------------------------------
__global__ void __launch_bounds__(256, 2)
k_diffusion(const bf16_t* __restrict__ S,
            const bf16_t* __restrict__ CT,
            bf16_t* __restrict__ XG,
            int J, int Cc, int colOff) {
  const int gw   = blockIdx.x * 8 + (threadIdx.x >> 5);
  const int lane = threadIdx.x & 31;
  const int lo = lane & 15, hi = lane >> 4;
  const int jt = J >> 6;                       // 64-wide col tiles (65 or 66)
  const int tn = gw / jt, tj = gw - tn * jt;
  const int n0 = tn * 64, j0 = tj * 64;
  v8f acc[4][4] = {};
  mma_tile64(S, NN, n0, CT, NN, j0, NN, lo, hi, acc);
#pragma unroll
  for (int s = 0; s < 4; ++s) {
    const int j = j0 + 16 * s + lo;
    const int bb = j / Cc, cc = j - bb * Cc;   // column (b,c) of this lane
#pragma unroll
    for (int r = 0; r < 4; ++r)
#pragma unroll
      for (int g = 0; g < 8; ++g) {
        int n = n0 + 16 * r + g + 8 * hi;
        XG[((size_t)(n * BB + bb)) * KP + colOff + cc] = tobf(acc[r][s][g]);
      }
  }
}

// ---------------------------------------------------------------------------
// 6) Dense GEMM + fused activation (64x64 tiles):
//    out[r,o] = act( sum_q XG[r][q] * WT[o][q] + bias[o] )
//    K = 224 (7 WMMA chunks). act: 0=sigmoid (gates z|r), 1=tanh (candidate).
// ---------------------------------------------------------------------------
__global__ void __launch_bounds__(256, 2)
k_dense(const bf16_t* __restrict__ XG,
        const bf16_t* __restrict__ WT,
        const float* __restrict__ bias,
        float* __restrict__ out, int O, int act) {
  const int gw   = blockIdx.x * 8 + (threadIdx.x >> 5);
  const int lane = threadIdx.x & 31;
  const int lo = lane & 15, hi = lane >> 4;
  const int ot = O >> 6;                       // 64-wide col tiles (2 or 1)
  const int tr = gw / ot, to = gw - tr * ot;
  const int r0 = tr * 64, o0 = to * 64;
  v8f acc[4][4] = {};
  mma_tile64(XG, KP, r0, WT, KP, o0, KP, lo, hi, acc);
#pragma unroll
  for (int s = 0; s < 4; ++s) {
    const int o = o0 + 16 * s + lo;
    const float bv = bias[o];
#pragma unroll
    for (int r = 0; r < 4; ++r)
#pragma unroll
      for (int g = 0; g < 8; ++g) {
        int row = r0 + 16 * r + g + 8 * hi;
        float v = acc[r][s][g] + bv;
        v = act ? tanhf(v) : sigmoidf_(v);
        out[(size_t)row * O + o] = v;
      }
  }
}

// ---------------------------------------------------------------------------
// 7) GRU blend: h = r*h + (1-r)*hc   (r = zr[:,64:128])
// ---------------------------------------------------------------------------
__global__ void k_hup(float* __restrict__ h, const float* __restrict__ zr,
                      const float* __restrict__ hc) {
  int idx = blockIdx.x * 256 + threadIdx.x;
  if (idx >= RTOT * HH) return;
  int r = idx >> 6, c = idx & 63;
  float rg = zr[(size_t)r * 128 + 64 + c];
  h[idx] = rg * h[idx] + (1.0f - rg) * hc[idx];
}

// ---------------------------------------------------------------------------
// 8) Projection: go[r] = h[r,:] . proj_W + proj_b ; also write final output.
// ---------------------------------------------------------------------------
__global__ void k_proj(const float* __restrict__ h, const float* __restrict__ pw,
                       const float* __restrict__ pb, float* __restrict__ go,
                       float* __restrict__ out, int t) {
  int r = blockIdx.x * 256 + threadIdx.x;
  if (r >= RTOT) return;
  int n = r >> 6, b = r & 63;
  float s = pb[0];
#pragma unroll
  for (int c = 0; c < HH; ++c) s += h[(size_t)r * HH + c] * pw[c];
  go[r] = s;
  out[((size_t)(b * HZ + t)) * NN + n] = s;   // [B, horizon, N, 1]
}

// ---------------------------------------------------------------------------
// Host-side launch sequence (graph-capture safe: only kernel launches).
// ---------------------------------------------------------------------------
extern "C" void kernel_launch(void* const* d_in, const int* in_sizes, int n_in,
                              void* d_out, int out_size, void* d_ws, size_t ws_size,
                              hipStream_t stream) {
  const float* x     = (const float*)d_in[0];   // [B,T,N,1]
  const float* ycov  = (const float*)d_in[1];   // [B,HZ,N,1]
  const float* emb   = (const float*)d_in[2];   // [N,8]
  const float* egW   = (const float*)d_in[3];   // [195,128]
  const float* egB   = (const float*)d_in[4];
  const float* euW   = (const float*)d_in[5];   // [195,64]
  const float* euB   = (const float*)d_in[6];
  const float* dgW   = (const float*)d_in[7];   // [198,128]
  const float* dgB   = (const float*)d_in[8];
  const float* duW   = (const float*)d_in[9];   // [198,64]
  const float* duB   = (const float*)d_in[10];
  const float* pW    = (const float*)d_in[11];  // [64,1]
  const float* pB    = (const float*)d_in[12];
  float* out = (float*)d_out;

  // deterministic workspace carve-up (256B aligned)
  char* ws = (char*)d_ws;
  auto carve = [&](size_t bytes) -> void* {
    void* p = ws; ws += (bytes + 255) & ~(size_t)255; return p;
  };
  bf16_t* S1b  = (bf16_t*)carve((size_t)NN * NN * 2);
  bf16_t* S1Tb = (bf16_t*)carve((size_t)NN * NN * 2);
  bf16_t* S2b  = (bf16_t*)carve((size_t)NN * NN * 2);
  bf16_t* WTeg = (bf16_t*)carve((size_t)128 * KP * 2);
  bf16_t* WTeu = (bf16_t*)carve((size_t)64  * KP * 2);
  bf16_t* WTdg = (bf16_t*)carve((size_t)128 * KP * 2);
  bf16_t* WTdu = (bf16_t*)carve((size_t)64  * KP * 2);
  float*  hbuf = (float*)carve((size_t)RTOT * HH * 4);
  bf16_t* CT   = (bf16_t*)carve((size_t)JDEC * NN * 2);
  bf16_t* XG   = (bf16_t*)carve((size_t)RTOT * KP * 2);
  float*  ZR   = (float*)carve((size_t)RTOT * 128 * 4);
  float*  HC   = (float*)carve((size_t)RTOT * HH * 4);
  float*  GO   = (float*)carve((size_t)RTOT * 4);

  // --- one-time setup ---
  k_adj <<<NN, 256, 0, stream>>>(emb, S1b, S1Tb);
  k_cheb<<<(16 * 16) / 8, 256, 0, stream>>>(S1b, S1Tb, S2b);
  k_wpack<<<(128 * KP + 255) / 256, 256, 0, stream>>>(egW, WTeg, 3 * CENC, 128);
  k_wpack<<<(64  * KP + 255) / 256, 256, 0, stream>>>(euW, WTeu, 3 * CENC, 64);
  k_wpack<<<(128 * KP + 255) / 256, 256, 0, stream>>>(dgW, WTdg, 3 * CDEC, 128);
  k_wpack<<<(64  * KP + 255) / 256, 256, 0, stream>>>(duW, WTdu, 3 * CDEC, 64);
  k_zero<<<(RTOT * HH) / 256, 256, 0, stream>>>(hbuf, RTOT * HH);
  k_zero<<<RTOT / 256, 256, 0, stream>>>(GO, RTOT);

  const int diffBlkE  = (16 * (JENC / 64)) / 8;   // 16*65/8 = 130, exact
  const int diffBlkD  = (16 * (JDEC / 64)) / 8;   // 16*66/8 = 132, exact
  const int denseBlkG = (1024 * 2) / 8;           // O=128 -> 256 blocks
  const int denseBlkU = (1024 * 1) / 8;           // O=64  -> 128 blocks

  // --- encoder ---
  for (int t = 0; t < TT; ++t) {
    k_build_enc<<<RTOT / 256, 256, 0, stream>>>(x, t, hbuf, ZR, 0, CT, XG);
    k_diffusion<<<diffBlkE, 256, 0, stream>>>(S1b, CT, XG, JENC, CENC, 1 * CENC);
    k_diffusion<<<diffBlkE, 256, 0, stream>>>(S2b, CT, XG, JENC, CENC, 2 * CENC);
    k_dense<<<denseBlkG, 256, 0, stream>>>(XG, WTeg, egB, ZR, 128, 0);   // zr=sigmoid
    k_build_enc<<<RTOT / 256, 256, 0, stream>>>(x, t, hbuf, ZR, 1, CT, XG);
    k_diffusion<<<diffBlkE, 256, 0, stream>>>(S1b, CT, XG, JENC, CENC, 1 * CENC);
    k_diffusion<<<diffBlkE, 256, 0, stream>>>(S2b, CT, XG, JENC, CENC, 2 * CENC);
    k_dense<<<denseBlkU, 256, 0, stream>>>(XG, WTeu, euB, HC, 64, 1);    // hc=tanh
    k_hup<<<(RTOT * HH) / 256, 256, 0, stream>>>(hbuf, ZR, HC);
  }

  // --- decoder (autoregressive) ---
  for (int t = 0; t < HZ; ++t) {
    k_build_dec<<<RTOT / 256, 256, 0, stream>>>(GO, ycov, t, hbuf, ZR, 0, CT, XG);
    k_diffusion<<<diffBlkD, 256, 0, stream>>>(S1b, CT, XG, JDEC, CDEC, 1 * CDEC);
    k_diffusion<<<diffBlkD, 256, 0, stream>>>(S2b, CT, XG, JDEC, CDEC, 2 * CDEC);
    k_dense<<<denseBlkG, 256, 0, stream>>>(XG, WTdg, dgB, ZR, 128, 0);
    k_build_dec<<<RTOT / 256, 256, 0, stream>>>(GO, ycov, t, hbuf, ZR, 1, CT, XG);
    k_diffusion<<<diffBlkD, 256, 0, stream>>>(S1b, CT, XG, JDEC, CDEC, 1 * CDEC);
    k_diffusion<<<diffBlkD, 256, 0, stream>>>(S2b, CT, XG, JDEC, CDEC, 2 * CDEC);
    k_dense<<<denseBlkU, 256, 0, stream>>>(XG, WTdu, duB, HC, 64, 1);
    k_hup<<<(RTOT * HH) / 256, 256, 0, stream>>>(hbuf, ZR, HC);
    k_proj<<<RTOT / 256, 256, 0, stream>>>(hbuf, pW, pB, GO, out, t);
  }
  (void)in_sizes; (void)n_in; (void)out_size; (void)ws_size;
}
